// HeteroGATEncoder_1176821039774
// MI455X (gfx1250) — matrix operand order repaired and checked by reading, not compiled, not run on hardware
//
#include <hip/hip_runtime.h>
#include <math.h>

#define NN 50000
#define DD 128
#define HH 4
#define LL 2
#define NEG 0.2f

typedef __attribute__((ext_vector_type(16))) _Float16 v16h;
typedef __attribute__((ext_vector_type(8)))  float    v8f;

static __device__ __forceinline__ float lrelu(float x) { return x > 0.f ? x : NEG * x; }

static __device__ __forceinline__ void atomicMaxF(float* addr, float val) {
    unsigned int* ua = (unsigned int*)addr;
    unsigned int old = *ua;
    while (__uint_as_float(old) < val) {
        unsigned int assumed = old;
        old = atomicCAS(ua, assumed, __float_as_uint(val));
        if (old == assumed) break;
    }
}

// ---------------- encoder: h = x @ enc_W + enc_b  (x: N x 4) ----------------
__global__ void k_encode(const float* __restrict__ x, const float* __restrict__ W,
                         const float* __restrict__ b, float* __restrict__ h) {
    int i = blockIdx.x * blockDim.x + threadIdx.x;
    if (i >= NN * DD) return;
    int n = i >> 7, d = i & 127;
    const float* xr = x + (size_t)n * 4;
    float acc = b[d];
    acc += xr[0] * W[0 * DD + d] + xr[1] * W[1 * DD + d] +
           xr[2] * W[2 * DD + d] + xr[3] * W[3 * DD + d];
    h[i] = acc;
}

// ---------------- generic f32 -> f16 convert ----------------
__global__ void k_f2h(const float* __restrict__ s, _Float16* __restrict__ d, int n) {
    int i = blockIdx.x * blockDim.x + threadIdx.x;
    if (i < n) d[i] = (_Float16)s[i];
}

// ---------------- WMMA GEMM: xl = h@Wl + bl, xr = h@Wr + br ----------------
// Block = 256 threads = 8 waves. Block handles 16 rows (M tile), wave w handles
// output columns [16w, 16w+16). K loop: 4 steps of 32 -> 8 WMMAs per wave.
// W held row-major (K x N) in f16; h row-major (M x K) in f16.
__global__ void k_gemm_xlxr(const _Float16* __restrict__ hF,
                            const _Float16* __restrict__ WlF,
                            const _Float16* __restrict__ WrF,
                            const float* __restrict__ bl, const float* __restrict__ br,
                            float* __restrict__ xl, float* __restrict__ xr) {
    const int lane = threadIdx.x & 31;
    const int wave = threadIdx.x >> 5;
    const int m0 = blockIdx.x * 16;
    const int n0 = wave * 16;

    // A (16x32 f16): lane = M row (lane&15); lanes 0-15 start K pairs at 0, lanes 16-31 at 8;
    // VGPR i: K = base + (i>>2)*16 + (i&3)*2  (per ISA 16-bit A table)
    const int arow = m0 + (lane & 15);
    const int akb  = (lane >> 4) * 8;
    // B (32x16 f16): lane = N col (lane&15); element s holds K = akb + (s>>3)*16 + (s&7) (same order as A)
    v8f accL = {};
    v8f accR = {};

    for (int k0 = 0; k0 < DD; k0 += 32) {
        union { v16h v; unsigned int u[8]; } A, BL, BR;
        const _Float16* ap = hF + (size_t)arow * DD + k0;
#pragma unroll
        for (int i = 0; i < 8; ++i) {
            int kk = akb + (i >> 2) * 16 + (i & 3) * 2;
            A.u[i] = *(const unsigned int*)(ap + kk);
        }
        const _Float16* blp = WlF + (size_t)k0 * DD + n0 + (lane & 15);
        const _Float16* brp = WrF + (size_t)k0 * DD + n0 + (lane & 15);
#pragma unroll
        for (int i = 0; i < 16; ++i) {
            int kk = akb + (i >> 3) * 16 + (i & 7);
            BL.v[i] = blp[(size_t)kk * DD];
            BR.v[i] = brp[(size_t)kk * DD];
        }
        accL = __builtin_amdgcn_wmma_f32_16x16x32_f16(false, A.v, false, BL.v,
                                                      (short)0, accL, false, false);
        accR = __builtin_amdgcn_wmma_f32_16x16x32_f16(false, A.v, false, BR.v,
                                                      (short)0, accR, false, false);
    }
    // D (16x16 f32): lane = N col (lane&15); VGPR i -> M = i + 8*(lane>=16)
    const int col = n0 + (lane & 15);
    const int mb  = m0 + (lane >> 4) * 8;
    const float bL = bl[col], bR = br[col];
#pragma unroll
    for (int i = 0; i < 8; ++i) {
        xl[(size_t)(mb + i) * DD + col] = accL[i] + bL;
        xr[(size_t)(mb + i) * DD + col] = accR[i] + bR;
    }
}

// ---------------- init segment-softmax state ----------------
__global__ void k_init_seg(float* __restrict__ amax, float* __restrict__ den) {
    int i = blockIdx.x * blockDim.x + threadIdx.x;
    if (i < NN * HH) { amax[i] = -3.0e38f; den[i] = 0.f; }
}

// ---------------- init out accumulator with summed biases ----------------
__global__ void k_init_out(const float* __restrict__ b0, const float* __restrict__ b1,
                           const float* __restrict__ b2, float* __restrict__ out) {
    int i = blockIdx.x * blockDim.x + threadIdx.x;
    if (i >= NN * DD) return;
    int d = i & 127;
    out[i] = b0[d] + b1[d] + b2[d];
}

// ---------------- pass 1: edge scores + segment max ----------------
// One wave per edge; lane covers 4 channels (head = lane>>3), butterfly reduce in groups of 8.
__global__ void k_score(const float* __restrict__ xl, const float* __restrict__ xr,
                        const int* __restrict__ src, const int* __restrict__ dst,
                        const float* __restrict__ att, const float* __restrict__ ea,
                        const float* __restrict__ We, float* __restrict__ a,
                        float* __restrict__ amax, int E) {
    int gw = (blockIdx.x * blockDim.x + threadIdx.x) >> 5;
    int lane = threadIdx.x & 31;
    if (gw >= E) return;
    int s = src[gw], d = dst[gw];
    const float4 lv = *(const float4*)(xl + (size_t)s * DD + lane * 4);
    const float4 rv = *(const float4*)(xr + (size_t)d * DD + lane * 4);
    const float4 av = *(const float4*)(att + lane * 4);
    float4 ev = make_float4(0.f, 0.f, 0.f, 0.f);
    if (We != nullptr) {
        float e = ea[gw];
        const float4 wv = *(const float4*)(We + lane * 4);
        ev = make_float4(e * wv.x, e * wv.y, e * wv.z, e * wv.w);
    }
    float part = lrelu(lv.x + rv.x + ev.x) * av.x
               + lrelu(lv.y + rv.y + ev.y) * av.y
               + lrelu(lv.z + rv.z + ev.z) * av.z
               + lrelu(lv.w + rv.w + ev.w) * av.w;
    part += __shfl_xor(part, 1);
    part += __shfl_xor(part, 2);
    part += __shfl_xor(part, 4);
    if ((lane & 7) == 0) {
        int hh = lane >> 3;
        a[(size_t)gw * HH + hh] = part;
        atomicMaxF(amax + (size_t)d * HH + hh, part);
    }
}

// ---------------- pass 2: exp + segment sum ----------------
__global__ void k_expsum(const int* __restrict__ dst, float* __restrict__ a,
                         const float* __restrict__ amax, float* __restrict__ den, int E) {
    int i = blockIdx.x * blockDim.x + threadIdx.x;
    if (i >= E * HH) return;
    int e = i >> 2, hh = i & 3;
    int d = dst[e];
    float v = expf(a[i] - amax[(size_t)d * HH + hh]);
    a[i] = v;
    atomicAdd(den + (size_t)d * HH + hh, v);
}

// ---------------- pass 3: alpha-weighted scatter add ----------------
__global__ void k_scatter(const float* __restrict__ xl, const int* __restrict__ src,
                          const int* __restrict__ dst, const float* __restrict__ a,
                          const float* __restrict__ den, float* __restrict__ out, int E) {
    int gw = (blockIdx.x * blockDim.x + threadIdx.x) >> 5;
    int lane = threadIdx.x & 31;
    if (gw >= E) return;
    int s = src[gw], d = dst[gw];
    int hh = lane >> 3;
    float alpha = a[(size_t)gw * HH + hh] / (den[(size_t)d * HH + hh] + 1e-16f);
    const float4 lv = *(const float4*)(xl + (size_t)s * DD + lane * 4);
    float* op = out + (size_t)d * DD + lane * 4;
    atomicAdd(op + 0, lv.x * alpha);
    atomicAdd(op + 1, lv.y * alpha);
    atomicAdd(op + 2, lv.z * alpha);
    atomicAdd(op + 3, lv.w * alpha);
}

// ---------------- ELU ----------------
__global__ void k_elu(const float* __restrict__ in, float* __restrict__ outp) {
    int i = blockIdx.x * blockDim.x + threadIdx.x;
    if (i >= NN * DD) return;
    float v = in[i];
    outp[i] = v > 0.f ? v : (expf(v) - 1.f);
}

extern "C" void kernel_launch(void* const* d_in, const int* in_sizes, int n_in,
                              void* d_out, int out_size, void* d_ws, size_t ws_size,
                              hipStream_t stream) {
    (void)n_in; (void)out_size; (void)ws_size;
    const float* x        = (const float*)d_in[0];
    const int*   ei_phys  = (const int*)d_in[1];
    const float* ea_phys  = (const float*)d_in[2];
    const int*   ei_align = (const int*)d_in[3];
    const int*   ei_logic = (const int*)d_in[4];
    const float* ea_logic = (const float*)d_in[5];
    const float* enc_W    = (const float*)d_in[6];
    const float* enc_b    = (const float*)d_in[7];

    // per-relation params (phys: 8..14, align: 15..20, logic: 21..27)
    const float* Wl[3]   = {(const float*)d_in[8],  (const float*)d_in[15], (const float*)d_in[21]};
    const float* bl[3]   = {(const float*)d_in[9],  (const float*)d_in[16], (const float*)d_in[22]};
    const float* Wr[3]   = {(const float*)d_in[10], (const float*)d_in[17], (const float*)d_in[23]};
    const float* br[3]   = {(const float*)d_in[11], (const float*)d_in[18], (const float*)d_in[24]};
    const float* att[3]  = {(const float*)d_in[12], (const float*)d_in[19], (const float*)d_in[25]};
    const float* bias[3] = {(const float*)d_in[13], (const float*)d_in[20], (const float*)d_in[26]};
    const float* We[3]   = {(const float*)d_in[14], nullptr,                (const float*)d_in[27]};

    const int E[3] = {in_sizes[2], in_sizes[3] / 2, in_sizes[5]};
    const int* srcp[3] = {ei_phys, ei_align, ei_logic};
    const int* dstp[3] = {ei_phys + E[0], ei_align + E[1], ei_logic + E[2]};
    const float* eap[3] = {ea_phys, nullptr, ea_logic};

    // ---------------- workspace carve-up ----------------
    char* p = (char*)d_ws;
    float*    h    = (float*)p;    p += (size_t)NN * DD * 4;
    _Float16* hF   = (_Float16*)p; p += (size_t)NN * DD * 2;
    float*    xl   = (float*)p;    p += (size_t)NN * DD * 4;
    float*    xr   = (float*)p;    p += (size_t)NN * DD * 4;
    float*    out  = (float*)p;    p += (size_t)NN * DD * 4;
    float*    aE   = (float*)p;    p += (size_t)800000 * HH * 4;
    float*    amax = (float*)p;    p += (size_t)NN * HH * 4;
    float*    den  = (float*)p;    p += (size_t)NN * HH * 4;
    _Float16* WlF[3]; _Float16* WrF[3];
    for (int r = 0; r < 3; ++r) {
        WlF[r] = (_Float16*)p; p += (size_t)LL * DD * DD * 2;
        WrF[r] = (_Float16*)p; p += (size_t)LL * DD * DD * 2;
    }

    const int T = 256;
    const int gND = (NN * DD + T - 1) / T;      // 25000
    const int gNH = (NN * HH + T - 1) / T;
    const int gW  = LL * DD * DD;               // 32768 elems per weight tensor

    // encoder
    k_encode<<<gND, T, 0, stream>>>(x, enc_W, enc_b, h);

    // convert all 6 weight tensors (both layers) to f16, row-major K x N
    for (int r = 0; r < 3; ++r) {
        k_f2h<<<(gW + T - 1) / T, T, 0, stream>>>(Wl[r], WlF[r], gW);
        k_f2h<<<(gW + T - 1) / T, T, 0, stream>>>(Wr[r], WrF[r], gW);
    }

    for (int l = 0; l < LL; ++l) {
        k_f2h<<<gND, T, 0, stream>>>(h, hF, NN * DD);
        k_init_out<<<gND, T, 0, stream>>>(bias[0] + l * DD, bias[1] + l * DD,
                                          bias[2] + l * DD, out);
        for (int r = 0; r < 3; ++r) {
            k_gemm_xlxr<<<NN / 16, T, 0, stream>>>(hF,
                    WlF[r] + (size_t)l * DD * DD, WrF[r] + (size_t)l * DD * DD,
                    bl[r] + l * DD, br[r] + l * DD, xl, xr);
            k_init_seg<<<gNH, T, 0, stream>>>(amax, den);
            int gE = (E[r] * 32 + T - 1) / T;   // one wave per edge
            k_score<<<gE, T, 0, stream>>>(xl, xr, srcp[r], dstp[r],
                    att[r] + l * DD, eap[r],
                    We[r] ? We[r] + l * DD : nullptr, aE, amax, E[r]);
            k_expsum<<<(E[r] * HH + T - 1) / T, T, 0, stream>>>(dstp[r], aE, amax, den, E[r]);
            k_scatter<<<gE, T, 0, stream>>>(xl, srcp[r], dstp[r], aE, den, out, E[r]);
        }
        float* dest = (l == LL - 1) ? (float*)d_out : h;
        k_elu<<<gND, T, 0, stream>>>(out, dest);
    }
}
